// DLRM_Net_15659450761728
// MI455X (gfx1250) — matrix operand order repaired and checked by reading, not compiled, not run on hardware
//
#include <hip/hip_runtime.h>
#include <math.h>

typedef __attribute__((ext_vector_type(16))) _Float16 v16h;
typedef __attribute__((ext_vector_type(2)))  _Float16 v2h;
typedef __attribute__((ext_vector_type(8)))  float    v8f;

#define BATCH 8192
#define NT    26
#define DIM   64
#define VROWS 200000
#define BAGL  10

// ---------------------------------------------------------------------------
// zero fill (u32 granularity)
// ---------------------------------------------------------------------------
__global__ void fill_zero_u32(unsigned int* __restrict__ p, long n) {
    long i = (long)blockIdx.x * blockDim.x + threadIdx.x;
    if (i < n) p[i] = 0u;
}

// ---------------------------------------------------------------------------
// fp32 [Nin][Kin] -> f16 [Nout][Kout], zero padded in both dims
// ---------------------------------------------------------------------------
__global__ void convert_pad_f16(const float* __restrict__ src,
                                _Float16* __restrict__ dst,
                                int Nin, int Kin, int Kout, long total) {
    long i = (long)blockIdx.x * blockDim.x + threadIdx.x;
    if (i >= total) return;
    int  k = (int)(i % Kout);
    long n = i / Kout;
    float v = (n < Nin && k < Kin) ? src[n * (long)Kin + k] : 0.0f;
    dst[i] = (_Float16)v;
}

// ---------------------------------------------------------------------------
// EmbeddingBag sum pooling. One wave32 per bag; each lane owns dims {2l,2l+1}
// so every embedding row is one coalesced 256B transaction (b64 per lane).
// fp32 accumulate, packed f16 (b32) store into padded interaction tensor
// T[b][32][64] at row (1+t).
// ---------------------------------------------------------------------------
__global__ __launch_bounds__(256) void embed_pool(const float* __restrict__ emb,
                                                  const int*   __restrict__ lSi,
                                                  _Float16*    __restrict__ T) {
    const int  lane = threadIdx.x & 31;
    const long bag  = (long)blockIdx.x * 8 + (threadIdx.x >> 5);
    const int  t    = (int)(bag / BATCH);
    const int  b    = (int)(bag % BATCH);
    const int*    idx  = lSi + (size_t)t * (BATCH * BAGL) + (size_t)b * BAGL;
    const float2* tab2 = (const float2*)(emb + (size_t)t * VROWS * DIM);
    float sx = 0.0f, sy = 0.0f;
#pragma unroll
    for (int l = 0; l < BAGL; ++l) {
        const float2 r = tab2[(size_t)idx[l] * 32 + lane];
        sx += r.x;
        sy += r.y;
    }
    v2h hv;
    hv.x = (_Float16)sx;
    hv.y = (_Float16)sy;
    *(v2h*)(T + ((size_t)b * 32 + 1 + t) * DIM + 2 * lane) = hv;
}

// ---------------------------------------------------------------------------
// WMMA GEMM, 16x64 output tile per wave (4 accumulators, A fragment reused 4x):
// Y[m][n] = relu(sum_k A[m][k]*W[n][k] + bias[n]).  K, N mult. of 32 / 64.
// Fragment layouts used here (wave32; A/B use contiguous k, same order in both):
//   A (16-bit 16x32):  lane l -> M = l&15 ; k = kbA + i,
//                      kbA = (l<16) ? 0 : 16
//   B (16-bit 32x16):  lane l -> N = l&15 ; k = kbB + i, kbB = (l<16) ? 0 : 16
//   C/D (f32 16x16):   lane l -> N = l&15 ; VGPR v -> M = v + ((l<16)?0:8)
// ---------------------------------------------------------------------------
__global__ __launch_bounds__(32) void gemm_relu_wmma4(
    const _Float16* __restrict__ A, const _Float16* __restrict__ W,
    const float* __restrict__ bias, int K,
    _Float16* __restrict__ out0, int ld0,
    _Float16* __restrict__ out1, int ld1) {
    const int lane = threadIdx.x;
    const int r0   = blockIdx.y * 16;
    const int n0   = blockIdx.x * 64;
    const int nn   = n0 + (lane & 15);
    const int m    = r0 + (lane & 15);
    const int kbA  = (lane < 16) ? 0 : 16;
    const int kbB  = (lane < 16) ? 0 : 16;
    const _Float16* arow = A + (size_t)m * K;
    const _Float16* wrow0 = W + (size_t)nn * K;

    v8f acc[4] = {};
    for (int k0 = 0; k0 < K; k0 += 32) {
        v16h af;
#pragma unroll
        for (int i = 0; i < 16; ++i)
            af[i] = arow[k0 + kbA + i];
#pragma unroll
        for (int t = 0; t < 4; ++t) {
            const _Float16* wrow = wrow0 + (size_t)(16 * t) * K;
            v16h bf;
#pragma unroll
            for (int i = 0; i < 16; ++i)
                bf[i] = wrow[k0 + kbB + i];
            acc[t] = __builtin_amdgcn_wmma_f32_16x16x32_f16(false, af, false, bf,
                                                            (short)0, acc[t], false, false);
        }
    }

    const int mbase = (lane < 16) ? 0 : 8;
#pragma unroll
    for (int t = 0; t < 4; ++t) {
        const int   n  = nn + 16 * t;
        const float bv = bias[n];
#pragma unroll
        for (int v = 0; v < 8; ++v) {
            float y = acc[t][v] + bv;
            y = (y > 0.0f) ? y : 0.0f;
            _Float16 h = (_Float16)y;
            out0[(size_t)(r0 + mbase + v) * ld0 + n] = h;
            if (out1) out1[(size_t)(r0 + mbase + v) * ld1 + n] = h;
        }
    }
}

// ---------------------------------------------------------------------------
// Final top layer: N padded 1->16, sigmoid, fp32 output. grid.x = B/16.
// ---------------------------------------------------------------------------
__global__ __launch_bounds__(32) void gemm_final_wmma(
    const _Float16* __restrict__ A, const _Float16* __restrict__ W,
    const float* __restrict__ bias, int K, float* __restrict__ out) {
    const int lane = threadIdx.x;
    const int r0   = blockIdx.x * 16;
    const int n    = lane & 15;
    const int m    = r0 + (lane & 15);
    const int kbA  = (lane < 16) ? 0 : 16;
    const int kbB  = (lane < 16) ? 0 : 16;
    const _Float16* arow = A + (size_t)m * K;
    const _Float16* wrow = W + (size_t)n * K;

    v8f acc = {};
    for (int k0 = 0; k0 < K; k0 += 32) {
        v16h af, bf;
#pragma unroll
        for (int i = 0; i < 16; ++i) {
            af[i] = arow[k0 + kbA + i];
            bf[i] = wrow[k0 + kbB + i];
        }
        acc = __builtin_amdgcn_wmma_f32_16x16x32_f16(false, af, false, bf,
                                                     (short)0, acc, false, false);
    }

    if (n == 0) {
        const float bv    = bias[0];
        const int   mbase = (lane < 16) ? 0 : 8;
#pragma unroll
        for (int v = 0; v < 8; ++v) {
            float y = acc[v] + bv;
            out[r0 + mbase + v] = 1.0f / (1.0f + __expf(-y));
        }
    }
}

// ---------------------------------------------------------------------------
// Dot interaction: per batch row, Z = Tp(32x64) * Tp^T (Tp rows 27..31 = 0).
// One wave per row; tiles (0,0),(1,0),(1,1) -> 6 WMMAs; scatter strict lower
// triangle into R[b][64 + i*(i-1)/2 + j] (f16).
// ---------------------------------------------------------------------------
__global__ __launch_bounds__(256) void interact_wmma(const _Float16* __restrict__ T,
                                                     _Float16* __restrict__ R) {
    const int lane = threadIdx.x & 31;
    const int b    = blockIdx.x * 8 + (threadIdx.x >> 5);
    const _Float16* Tb = T + (size_t)b * 2048;   // [32][64]
    const int j    = lane & 15;
    const int kbA  = (lane < 16) ? 0 : 16;
    const int kbB  = (lane < 16) ? 0 : 16;

    v8f a00 = {}, a10 = {}, a11 = {};
    for (int k0 = 0; k0 < 64; k0 += 32) {
        v16h A0, A1, B0, B1;
#pragma unroll
        for (int i = 0; i < 16; ++i) {
            int ka = k0 + kbA + i;
            A0[i] = Tb[(size_t)j * 64 + ka];
            A1[i] = Tb[(size_t)(16 + j) * 64 + ka];
            B0[i] = Tb[(size_t)j * 64 + k0 + kbB + i];
            B1[i] = Tb[(size_t)(16 + j) * 64 + k0 + kbB + i];
        }
        a00 = __builtin_amdgcn_wmma_f32_16x16x32_f16(false, A0, false, B0, (short)0, a00, false, false);
        a10 = __builtin_amdgcn_wmma_f32_16x16x32_f16(false, A1, false, B0, (short)0, a10, false, false);
        a11 = __builtin_amdgcn_wmma_f32_16x16x32_f16(false, A1, false, B1, (short)0, a11, false, false);
    }

    _Float16* Rb    = R + (size_t)b * 416;
    const int mbase = (lane < 16) ? 0 : 8;
#pragma unroll
    for (int v = 0; v < 8; ++v) {
        int i0 = mbase + v;                     // tile (0,0): i in 0..15, j in 0..15
        if (i0 < 27 && j < i0)
            Rb[64 + i0 * (i0 - 1) / 2 + j] = (_Float16)a00[v];
        int i1 = 16 + mbase + v;                // tiles (1,0),(1,1): i in 16..31
        if (i1 < 27) {
            Rb[64 + i1 * (i1 - 1) / 2 + j] = (_Float16)a10[v];   // j<16<=i1 always
            int j1 = 16 + j;
            if (j1 < i1)
                Rb[64 + i1 * (i1 - 1) / 2 + j1] = (_Float16)a11[v];
        }
    }
}

// ---------------------------------------------------------------------------
extern "C" void kernel_launch(void* const* d_in, const int* in_sizes, int n_in,
                              void* d_out, int out_size, void* d_ws, size_t ws_size,
                              hipStream_t stream) {
    const float* x     = (const float*)d_in[0];
    const int*   lSi   = (const int*)  d_in[1];
    // d_in[2] = lS_o (unused: fixed bag length)
    const float* emb   = (const float*)d_in[3];
    const float* botW1 = (const float*)d_in[4];
    const float* botb1 = (const float*)d_in[5];
    const float* botW2 = (const float*)d_in[6];
    const float* botb2 = (const float*)d_in[7];
    const float* botW3 = (const float*)d_in[8];
    const float* botb3 = (const float*)d_in[9];
    const float* topW1 = (const float*)d_in[10];
    const float* topb1 = (const float*)d_in[11];
    const float* topW2 = (const float*)d_in[12];
    const float* topb2 = (const float*)d_in[13];
    const float* topW3 = (const float*)d_in[14];
    const float* topb3 = (const float*)d_in[15];
    float* out = (float*)d_out;

    // ---- workspace layout (f16 elements) ----
    _Float16* Xpad = (_Float16*)d_ws;                       // 8192*32
    _Float16* W1b  = Xpad + (size_t)BATCH * 32;             // 512*32
    _Float16* W2b  = W1b  + (size_t)512 * 32;               // 256*512
    _Float16* W3b  = W2b  + (size_t)256 * 512;              // 64*256
    _Float16* W1t  = W3b  + (size_t)64 * 256;               // 512*416
    _Float16* W2t  = W1t  + (size_t)512 * 416;              // 256*512
    _Float16* W3t  = W2t  + (size_t)256 * 512;              // 16*256
    _Float16* act1 = W3t  + (size_t)16 * 256;               // 8192*512 (also t1)
    _Float16* act2 = act1 + (size_t)BATCH * 512;            // 8192*256 (also t2)
    _Float16* Tm   = act2 + (size_t)BATCH * 256;            // 8192*32*64
    _Float16* Rm   = Tm   + (size_t)BATCH * 2048;           // 8192*416

    const int CB = 256;

    // zero-init padded interaction tensor T and feature matrix R
    {
        long n = (long)BATCH * 2048 / 2;  // u32 count
        fill_zero_u32<<<(unsigned)((n + CB - 1) / CB), CB, 0, stream>>>((unsigned*)Tm, n);
    }
    {
        long n = (long)BATCH * 416 / 2;
        fill_zero_u32<<<(unsigned)((n + CB - 1) / CB), CB, 0, stream>>>((unsigned*)Rm, n);
    }

    // weight / input conversion to f16 with zero padding
    {
        long t = (long)BATCH * 32;
        convert_pad_f16<<<(unsigned)((t + CB - 1) / CB), CB, 0, stream>>>(x, Xpad, BATCH, 13, 32, t);
    }
    {
        long t = 512 * 32;
        convert_pad_f16<<<(unsigned)((t + CB - 1) / CB), CB, 0, stream>>>(botW1, W1b, 512, 13, 32, t);
    }
    {
        long t = 256 * 512;
        convert_pad_f16<<<(unsigned)((t + CB - 1) / CB), CB, 0, stream>>>(botW2, W2b, 256, 512, 512, t);
    }
    {
        long t = 64 * 256;
        convert_pad_f16<<<(unsigned)((t + CB - 1) / CB), CB, 0, stream>>>(botW3, W3b, 64, 256, 256, t);
    }
    {
        long t = 512 * 416;
        convert_pad_f16<<<(unsigned)((t + CB - 1) / CB), CB, 0, stream>>>(topW1, W1t, 512, 415, 416, t);
    }
    {
        long t = 256 * 512;
        convert_pad_f16<<<(unsigned)((t + CB - 1) / CB), CB, 0, stream>>>(topW2, W2t, 256, 512, 512, t);
    }
    {
        long t = 16 * 256;
        convert_pad_f16<<<(unsigned)((t + CB - 1) / CB), CB, 0, stream>>>(topW3, W3t, 1, 256, 256, t);
    }

    // embedding bag sum pooling -> T rows 1..26 (one wave per bag)
    embed_pool<<<(NT * BATCH) / 8, 256, 0, stream>>>(emb, lSi, Tm);

    // bottom MLP (16x64 tile per wave)
    gemm_relu_wmma4<<<dim3(512 / 64, BATCH / 16), 32, 0, stream>>>(
        Xpad, W1b, botb1, 32, act1, 512, (_Float16*)nullptr, 0);
    gemm_relu_wmma4<<<dim3(256 / 64, BATCH / 16), 32, 0, stream>>>(
        act1, W2b, botb2, 512, act2, 256, (_Float16*)nullptr, 0);
    // layer 3 writes h into T[b][0][:] (row stride 2048) and R[b][0:64] (stride 416)
    gemm_relu_wmma4<<<dim3(64 / 64, BATCH / 16), 32, 0, stream>>>(
        act2, W3b, botb3, 256, Tm, 2048, Rm, 416);

    // dot interaction -> R[b][64:415]
    interact_wmma<<<BATCH / 8, 256, 0, stream>>>(Tm, Rm);

    // top MLP (act1/act2 reused as t1/t2)
    gemm_relu_wmma4<<<dim3(512 / 64, BATCH / 16), 32, 0, stream>>>(
        Rm, W1t, topb1, 416, act1, 512, (_Float16*)nullptr, 0);
    gemm_relu_wmma4<<<dim3(256 / 64, BATCH / 16), 32, 0, stream>>>(
        act1, W2t, topb2, 512, act2, 256, (_Float16*)nullptr, 0);
    gemm_final_wmma<<<BATCH / 16, 32, 0, stream>>>(act2, W3t, topb3, 256, out);
}
